// Respiso_77000173683172
// MI455X (gfx1250) — matrix operand order repaired and checked by reading, not compiled, not run on hardware
//
#include <hip/hip_runtime.h>
#include <hip/hip_bf16.h>

// ---------------------------------------------------------------------------
// Problem constants (from reference)
// ---------------------------------------------------------------------------
#define VOCAB 32000
#define DMODEL 512
#define BATCH 8
#define SEQ 1024
#define MROWS (BATCH * SEQ)      // 8192
#define LN_EPS 1e-5f

typedef __bf16 bf16_t;
typedef __attribute__((ext_vector_type(16))) __bf16 v16bf;
typedef __attribute__((ext_vector_type(8)))  float  v8f;
typedef __attribute__((ext_vector_type(4)))  float  v4f;

// ---------------------------------------------------------------------------
// Helpers
// ---------------------------------------------------------------------------
__device__ __forceinline__ bf16_t f2bf(float x) {
    unsigned u = __float_as_uint(x);
    unsigned r = (u + 0x7FFFu + ((u >> 16) & 1u)) >> 16;   // round-to-nearest-even
    union { unsigned short s; bf16_t b; } cv;
    cv.s = (unsigned short)r;
    return cv.b;
}

// A fragment (16x32 bf16, MxK): lane holds row (lane&15); K values
// [base, base+16) contiguous with base = 16*(lane>>4).
// p points at row_ptr + base; two 16-byte loads.
__device__ __forceinline__ v16bf load_frag_a(const bf16_t* p) {
    union { v16bf v; struct { v4f lo; v4f hi; } q; } u;
    u.q.lo = *(const v4f*)(p);
    u.q.hi = *(const v4f*)(p + 8);
    return u.v;
}

// B fragment (32x16 bf16, KxN): lane holds column (lane&15); K values
// [base, base+16) contiguous with base = 16*(lane>>4).  Two 16-byte loads
// (the B operand source is K-contiguous: W^T rows / E rows).
__device__ __forceinline__ v16bf load_frag_b(const bf16_t* p) {
    union { v16bf v; struct { v4f lo; v4f hi; } q; } u;
    u.q.lo = *(const v4f*)(p);
    u.q.hi = *(const v4f*)(p + 8);
    return u.v;
}

__device__ __forceinline__ v8f wmma_bf16(v16bf a, v16bf b, v8f c) {
    // (neg_a, A, neg_b, B, c_mod, C, reuse_a, reuse_b)
    return __builtin_amdgcn_wmma_f32_16x16x32_bf16(false, a, false, b,
                                                   (short)0, c, false, false);
}

// ---------------------------------------------------------------------------
// K0a: E fp32 -> bf16   (V*D = 16,384,000 elements, vectorized x4)
// ---------------------------------------------------------------------------
__global__ void convert_e_kernel(const float* __restrict__ E,
                                 bf16_t* __restrict__ Ebf) {
    size_t i = ((size_t)blockIdx.x * blockDim.x + threadIdx.x) * 4;
    if (i >= (size_t)VOCAB * DMODEL) return;
    float4 v = *(const float4*)(E + i);
    Ebf[i + 0] = f2bf(v.x);
    Ebf[i + 1] = f2bf(v.y);
    Ebf[i + 2] = f2bf(v.z);
    Ebf[i + 3] = f2bf(v.w);
}

// ---------------------------------------------------------------------------
// K0b: transpose + convert the three DxD weights: Wt[z][n][k] = (bf16)W[z][k][n]
// LDS-tiled so both global read and write are coalesced.
// ---------------------------------------------------------------------------
__global__ void transpose_w_kernel(const float* __restrict__ W0,
                                   const float* __restrict__ W1,
                                   const float* __restrict__ W2,
                                   bf16_t* __restrict__ Wt) {
    __shared__ float tile[16][17];
    const float* W = (blockIdx.z == 0) ? W0 : (blockIdx.z == 1) ? W1 : W2;
    bf16_t* out = Wt + (size_t)blockIdx.z * DMODEL * DMODEL;
    int kb = blockIdx.x * 16, nb = blockIdx.y * 16;
    int tx = threadIdx.x, ty = threadIdx.y;
    tile[ty][tx] = W[(size_t)(kb + ty) * DMODEL + nb + tx];
    __syncthreads();
    out[(size_t)(nb + ty) * DMODEL + kb + tx] = f2bf(tile[tx][ty]);
}

// ---------------------------------------------------------------------------
// K1: fused embedding-gather + 3 GEMMs + gate activations.
//   f    = sigmoid(emb @ W_ih + b_h)
//   xmix = silu(emb @ W1 + b1) * (emb @ W2 + b2)
// A rows come straight from Ebf[tokens[m]] (no materialized emb).
// Block = 256 thr (8 waves stacked in M) -> 128x64 tile; grid 64 x 8.
// All 12 B fragments of a K-step are batch-loaded into distinct registers
// before any WMMA issues: one s_wait per K-step, not one per WMMA.
// ---------------------------------------------------------------------------
__global__ void gate_gemm_kernel(const int* __restrict__ tokens,
                                 const bf16_t* __restrict__ Ebf,
                                 const bf16_t* __restrict__ Wt,   // 3 packed D*D
                                 const float* __restrict__ b_h,
                                 const float* __restrict__ b1,
                                 const float* __restrict__ b2,
                                 float* __restrict__ fout,
                                 float* __restrict__ xmout) {
    const int lane = threadIdx.x & 31;
    const int wave = threadIdx.x >> 5;
    const int lm = lane & 15;
    const int hi = lane >> 4;
    const int m0 = blockIdx.x * 128 + wave * 16;
    const int n0 = blockIdx.y * 64;

    const int tok = tokens[m0 + lm];
    const bf16_t* arow = Ebf + (size_t)tok * DMODEL + hi * 16;
    const bf16_t* wt0 = Wt;

    v8f acc[3][4];
    v8f zero = {};
#pragma unroll
    for (int z = 0; z < 3; ++z)
#pragma unroll
        for (int nt = 0; nt < 4; ++nt) acc[z][nt] = zero;

    for (int kk = 0; kk < DMODEL; kk += 32) {
        v16bf af = load_frag_a(arow + kk);
        __builtin_prefetch(arow + kk + 32, 0, 3);
        v16bf bf[3][4];
#pragma unroll
        for (int nt = 0; nt < 4; ++nt) {
            size_t boff = (size_t)(n0 + nt * 16 + lm) * DMODEL + kk + hi * 16;
#pragma unroll
            for (int z = 0; z < 3; ++z)
                bf[z][nt] = load_frag_b(wt0 + (size_t)z * DMODEL * DMODEL + boff);
        }
#pragma unroll
        for (int nt = 0; nt < 4; ++nt)
#pragma unroll
            for (int z = 0; z < 3; ++z)
                acc[z][nt] = wmma_bf16(af, bf[z][nt], acc[z][nt]);
    }

#pragma unroll
    for (int nt = 0; nt < 4; ++nt) {
        int n = n0 + nt * 16 + lm;
        float bh = b_h[n], bb1 = b1[n], bb2 = b2[n];
#pragma unroll
        for (int r = 0; r < 8; ++r) {
            int m = m0 + r + hi * 8;
            float gi = acc[0][nt][r] + bh;
            float fv = 1.0f / (1.0f + __expf(-gi));
            float z1 = acc[1][nt][r] + bb1;
            float z2 = acc[2][nt][r] + bb2;
            float xm = (z1 / (1.0f + __expf(-z1))) * z2;   // silu(z1)*z2
            size_t o = (size_t)m * DMODEL + n;
            fout[o]  = fv;
            xmout[o] = xm;
        }
    }
}

// ---------------------------------------------------------------------------
// K2: sequential scan.  One wave32 per batch element; each lane owns 16 dims
// (d = lane + 32*j, coalesced).  Both LayerNorms per step are pure intra-wave
// shuffle reductions (no barriers).  The next step's f/x loads are issued
// before the current step's math (register pipeline) and the step after that
// is prefetched, hiding global latency behind the serial recurrence.
// ---------------------------------------------------------------------------
__global__ void scan_kernel(const float* __restrict__ f,
                            const float* __restrict__ x,
                            const float* __restrict__ ln_g,
                            const float* __restrict__ ln_b,
                            const float* __restrict__ h0,
                            const float* __restrict__ lnf_g,
                            const float* __restrict__ lnf_b,
                            bf16_t* __restrict__ xout) {
    const int b = blockIdx.x;
    const int lane = threadIdx.x;           // 0..31
    const float inv_d = 1.0f / (float)DMODEL;

    float h[16], g[16], bt[16], fg[16], fb[16];
#pragma unroll
    for (int j = 0; j < 16; ++j) {
        int d = lane + 32 * j;
        h[j]  = h0[d];
        g[j]  = ln_g[d];
        bt[j] = ln_b[d];
        fg[j] = lnf_g[d];
        fb[j] = lnf_b[d];
    }

    // Preload step 0.
    float fv[16], xv[16];
    {
        size_t base0 = (size_t)b * SEQ * DMODEL;
#pragma unroll
        for (int j = 0; j < 16; ++j) {
            int d = lane + 32 * j;
            fv[j] = f[base0 + d];
            xv[j] = x[base0 + d];
        }
    }

    for (int s = 0; s < SEQ; ++s) {
        size_t base = ((size_t)b * SEQ + s) * DMODEL;

        // Issue next step's loads before any math on this step.
        float fvN[16], xvN[16];
        if (s + 1 < SEQ) {
            size_t baseN = base + DMODEL;
#pragma unroll
            for (int j = 0; j < 16; ++j) {
                int d = lane + 32 * j;
                fvN[j] = f[baseN + d];
                xvN[j] = x[baseN + d];
            }
            __builtin_prefetch(f + baseN + DMODEL + lane, 0, 3);
            __builtin_prefetch(x + baseN + DMODEL + lane, 0, 3);
        }

        float r[16];
        float sum = 0.0f, sq = 0.0f;
#pragma unroll
        for (int j = 0; j < 16; ++j) {
            float hr = fv[j] * h[j] + (1.0f - fv[j]) * xv[j];
            r[j] = hr;
            sum += hr;
            sq  += hr * hr;
        }
#pragma unroll
        for (int o = 16; o > 0; o >>= 1) {
            sum += __shfl_xor(sum, o, 32);
            sq  += __shfl_xor(sq,  o, 32);
        }
        float mean = sum * inv_d;
        float rstd = rsqrtf(sq * inv_d - mean * mean + LN_EPS);

        float sum2 = 0.0f, sq2 = 0.0f;
#pragma unroll
        for (int j = 0; j < 16; ++j) {
            float hn = (r[j] - mean) * rstd * g[j] + bt[j] + h[j];
            h[j] = hn;
            sum2 += hn;
            sq2  += hn * hn;
        }
#pragma unroll
        for (int o = 16; o > 0; o >>= 1) {
            sum2 += __shfl_xor(sum2, o, 32);
            sq2  += __shfl_xor(sq2,  o, 32);
        }
        float mean2 = sum2 * inv_d;
        float rstd2 = rsqrtf(sq2 * inv_d - mean2 * mean2 + LN_EPS);

#pragma unroll
        for (int j = 0; j < 16; ++j) {
            int d = lane + 32 * j;
            float xo = (h[j] - mean2) * rstd2 * fg[j] + fb[j];
            xout[base + d] = f2bf(xo);
        }

        if (s + 1 < SEQ) {
#pragma unroll
            for (int j = 0; j < 16; ++j) { fv[j] = fvN[j]; xv[j] = xvN[j]; }
        }
    }
}

// ---------------------------------------------------------------------------
// K3: logits = X @ E^T  (M=8192, N=32000, K=512), bf16 WMMA, f32 out.
// Block = 256 thr (8 waves stacked in M) -> 128x128 tile; grid 64 x 250.
// Fully software-pipelined: A + all 8 B fragments are double-buffered in
// registers, so iteration k+1's 18 global_load_b128 issue before iteration
// k's 8 WMMAs execute — the loadcnt wait covers a full K-step of latency.
// ---------------------------------------------------------------------------
__global__ void logits_gemm_kernel(const bf16_t* __restrict__ Xbf,
                                   const bf16_t* __restrict__ Ebf,
                                   float* __restrict__ out) {
    const int lane = threadIdx.x & 31;
    const int wave = threadIdx.x >> 5;
    const int lm = lane & 15;
    const int hi = lane >> 4;
    const int m0 = blockIdx.x * 128 + wave * 16;
    const int n0 = blockIdx.y * 128;

    const bf16_t* arow = Xbf + (size_t)(m0 + lm) * DMODEL + hi * 16;
    const bf16_t* brow = Ebf + (size_t)(n0 + lm) * DMODEL + hi * 16;

    v8f acc[8];
    v8f zero = {};
#pragma unroll
    for (int nt = 0; nt < 8; ++nt) acc[nt] = zero;

    // Prologue: load K-step 0 fragments.
    v16bf aCur = load_frag_a(arow);
    v16bf bCur[8];
#pragma unroll
    for (int nt = 0; nt < 8; ++nt)
        bCur[nt] = load_frag_b(brow + (size_t)nt * 16 * DMODEL);

    for (int kk = 0; kk < DMODEL; kk += 32) {
        const int kn = (kk + 32 < DMODEL) ? kk + 32 : 0;   // last-iter reload, unused
        v16bf aNxt = load_frag_a(arow + kn);
        v16bf bNxt[8];
#pragma unroll
        for (int nt = 0; nt < 8; ++nt)
            bNxt[nt] = load_frag_b(brow + (size_t)nt * 16 * DMODEL + kn);

#pragma unroll
        for (int nt = 0; nt < 8; ++nt)
            acc[nt] = wmma_bf16(aCur, bCur[nt], acc[nt]);

        aCur = aNxt;
#pragma unroll
        for (int nt = 0; nt < 8; ++nt) bCur[nt] = bNxt[nt];
    }

#pragma unroll
    for (int nt = 0; nt < 8; ++nt) {
        int n = n0 + nt * 16 + lm;
#pragma unroll
        for (int r = 0; r < 8; ++r) {
            int m = m0 + r + hi * 8;
            out[(size_t)m * VOCAB + n] = acc[nt][r];
        }
    }
}

// ---------------------------------------------------------------------------
// Launcher
// ---------------------------------------------------------------------------
extern "C" void kernel_launch(void* const* d_in, const int* in_sizes, int n_in,
                              void* d_out, int out_size, void* d_ws, size_t ws_size,
                              hipStream_t stream) {
    const int*   tokens = (const int*)  d_in[0];
    const float* E      = (const float*)d_in[1];
    const float* W_ih   = (const float*)d_in[2];
    const float* b_h    = (const float*)d_in[3];
    const float* W1     = (const float*)d_in[4];
    const float* b1     = (const float*)d_in[5];
    const float* W2     = (const float*)d_in[6];
    const float* b2     = (const float*)d_in[7];
    const float* ln_g   = (const float*)d_in[8];
    const float* ln_b   = (const float*)d_in[9];
    const float* h0     = (const float*)d_in[10];
    const float* lnf_g  = (const float*)d_in[11];
    const float* lnf_b  = (const float*)d_in[12];
    float* out = (float*)d_out;

    // Workspace carve-up (all offsets 256B-aligned).
    char* ws = (char*)d_ws;
    const size_t sz_Ebf = (size_t)VOCAB * DMODEL * sizeof(bf16_t);   // 32,768,000
    const size_t sz_Wt  = 3ull * DMODEL * DMODEL * sizeof(bf16_t);   //  1,572,864
    const size_t sz_Xbf = (size_t)MROWS * DMODEL * sizeof(bf16_t);   //  8,388,608
    const size_t sz_f   = (size_t)MROWS * DMODEL * sizeof(float);    // 16,777,216
    bf16_t* Ebf = (bf16_t*)(ws);
    bf16_t* Wt  = (bf16_t*)(ws + sz_Ebf);
    bf16_t* Xbf = (bf16_t*)(ws + sz_Ebf + sz_Wt);
    float*  fbuf= (float*) (ws + sz_Ebf + sz_Wt + sz_Xbf);
    float*  xm  = (float*) (ws + sz_Ebf + sz_Wt + sz_Xbf + sz_f);

    // K0a: E -> bf16 (16,384,000 elems / 4 per thread / 256 per block)
    convert_e_kernel<<<16000, 256, 0, stream>>>(E, Ebf);

    // K0b: W_ih/W1/W2 -> transposed bf16
    transpose_w_kernel<<<dim3(32, 32, 3), dim3(16, 16), 0, stream>>>(W_ih, W1, W2, Wt);

    // K1: gather + 3 GEMMs + activations
    gate_gemm_kernel<<<dim3(MROWS / 128, DMODEL / 64), 256, 0, stream>>>(
        tokens, Ebf, Wt, b_h, b1, b2, fbuf, xm);

    // K2: recurrence + both LayerNorms, bf16 output
    scan_kernel<<<BATCH, 32, 0, stream>>>(fbuf, xm, ln_g, ln_b, h0, lnf_g, lnf_b, Xbf);

    // K3: logits = X @ E^T
    logits_gemm_kernel<<<dim3(MROWS / 128, VOCAB / 128), 256, 0, stream>>>(Xbf, Ebf, out);
}
